// PointNetFeaturePropagation_19061064860309
// MI455X (gfx1250) — compile-verified, matrix operand order checked by reading, not run on hardware
//
#include <hip/hip_runtime.h>

// ---------------------------------------------------------------------------
// Types for CDNA5 WMMA (wave32): v_wmma_f32_16x16x32_bf16
// ---------------------------------------------------------------------------
typedef __attribute__((ext_vector_type(16))) __bf16 v16bf;
typedef __attribute__((ext_vector_type(8)))  __bf16 v8bf;
typedef __attribute__((ext_vector_type(8)))  float  v8f;

#define DIST_EPS 1e-8f
#define BN_EPS   1e-5f

// Problem constants (from reference setup_inputs)
#define PB   8
#define PN   8192
#define PS   2048
#define PC1  128
#define PC2  256
#define PK1  384          // C1 + C2
#define PO1  256
#define PO2  128
#define PM   (PB * PN)    // 65536 rows

// ---------------------------------------------------------------------------
// fp32 -> bf16 conversion (for weights)
// ---------------------------------------------------------------------------
__global__ void fp_f32_to_bf16(const float* __restrict__ in,
                               __bf16* __restrict__ out, int n) {
  int i = blockIdx.x * blockDim.x + threadIdx.x;
  if (i < n) out[i] = (__bf16)in[i];
}

__global__ void fp_zero_f32(float* __restrict__ p, int n) {
  int i = blockIdx.x * blockDim.x + threadIdx.x;
  if (i < n) p[i] = 0.0f;
}

// ---------------------------------------------------------------------------
// 3-NN interpolation + concat.  One block = 256 query points of one batch.
// xyz2 for the batch (24 KB) staged into LDS with CDNA5 async global->LDS
// copies (GLOBAL_LOAD_ASYNC_TO_LDS_B128, tracked by ASYNCcnt).
// Writes X1[row, 0:128]   = points1 row (bf16)
//        X1[row, 128:384] = 3-NN weighted interp of points2 (bf16)
// ---------------------------------------------------------------------------
__global__ __launch_bounds__(256)
void fp_interp_concat(const float* __restrict__ xyz1,
                      const float* __restrict__ xyz2,
                      const float* __restrict__ points1,
                      const float* __restrict__ points2,
                      __bf16* __restrict__ X1) {
  __shared__ float sq[PS * 3];   // interleaved x,y,z : 24576 bytes
  const int b = blockIdx.y;
  const int n = blockIdx.x * 256 + threadIdx.x;

  // ---- async stage xyz2[b] into LDS: 256 threads x 6 x 16B = 24576 B ----
  {
    const char*    src      = (const char*)(xyz2 + (size_t)b * PS * 3);
    const unsigned lds_base = (unsigned)(uintptr_t)(void*)sq;  // LDS addr = low 32 bits
    #pragma unroll
    for (int i = 0; i < 6; ++i) {
      const unsigned dst = lds_base + threadIdx.x * 16u + (unsigned)i * 4096u;
      const unsigned long long ga =
          (unsigned long long)(uintptr_t)(src + threadIdx.x * 16u + (unsigned)i * 4096u);
      asm volatile("global_load_async_to_lds_b128 %0, %1, off"
                   :: "v"(dst), "v"(ga)
                   : "memory");
    }
    asm volatile("s_wait_asynccnt 0" ::: "memory");
  }
  __syncthreads();

  const size_t row = (size_t)b * PN + n;
  const float* p = xyz1 + row * 3;
  const float px = p[0], py = p[1], pz = p[2];

  float bd0 = 3.4e38f, bd1 = 3.4e38f, bd2 = 3.4e38f;
  int   bi0 = 0, bi1 = 0, bi2 = 0;
  for (int s = 0; s < PS; ++s) {
    const float dx = px - sq[3 * s + 0];   // uniform address -> LDS broadcast
    const float dy = py - sq[3 * s + 1];
    const float dz = pz - sq[3 * s + 2];
    const float d2 = dx * dx + dy * dy + dz * dz;
    if (d2 < bd2) {
      if (d2 < bd0)      { bd2 = bd1; bi2 = bi1; bd1 = bd0; bi1 = bi0; bd0 = d2; bi0 = s; }
      else if (d2 < bd1) { bd2 = bd1; bi2 = bi1; bd1 = d2;  bi1 = s; }
      else               { bd2 = d2;  bi2 = s; }
    }
  }

  const float r0 = 1.0f / (sqrtf(fmaxf(bd0, 0.0f)) + DIST_EPS);
  const float r1 = 1.0f / (sqrtf(fmaxf(bd1, 0.0f)) + DIST_EPS);
  const float r2 = 1.0f / (sqrtf(fmaxf(bd2, 0.0f)) + DIST_EPS);
  const float rs = 1.0f / (r0 + r1 + r2);
  const float w0 = r0 * rs, w1 = r1 * rs, w2 = r2 * rs;

  __bf16* xrow = X1 + row * PK1;
  const float* p1row = points1 + row * PC1;
  #pragma unroll 4
  for (int c = 0; c < PC1; ++c) xrow[c] = (__bf16)p1row[c];

  const float* p2b = points2 + (size_t)b * PS * PC2;
  const float* g0 = p2b + (size_t)bi0 * PC2;
  const float* g1 = p2b + (size_t)bi1 * PC2;
  const float* g2 = p2b + (size_t)bi2 * PC2;
  #pragma unroll 4
  for (int c = 0; c < PC2; ++c)
    xrow[PC1 + c] = (__bf16)(w0 * g0[c] + w1 * g1[c] + w2 * g2[c]);
}

// ---------------------------------------------------------------------------
// WMMA GEMM:  Y[M, Cout] = X[M, K] * W[Cout, K]^T + bias     (bf16 -> f32)
// One wave computes a 16-row strip across NT 16-wide N tiles (NT*8 acc VGPRs);
// the A fragment is loaded once per K-step and reused for all NT WMMAs.
// K is a template parameter so B-tile strides fold into instruction offsets.
//
// A fragment (16x32 bf16): lane l<16  -> row tileM+l,    K {k..k+7,  k+16..k+23}
//                          lane l>=16 -> row tileM+l-16, K {k+8..k+15,k+24..k+31}
// B fragment mirrors A with "row" = output channel (B = W^T).
// C/D (16x16 f32, 8 VGPRs): vgpr r, lane l -> row tileM + r + 8*(l>=16),
//                                            col tileN + (l&15).
// ---------------------------------------------------------------------------
template <int NT, int K>
__global__ __launch_bounds__(256)
void fp_gemm_bf16_wmma(const __bf16* __restrict__ X,
                       const __bf16* __restrict__ W,
                       const float* __restrict__ bias,
                       float* __restrict__ Y,
                       int M, int Cout) {
  const int lane   = threadIdx.x & 31;
  const int wave   = threadIdx.x >> 5;
  const int half   = lane >> 4;   // 0 = lanes 0-15, 1 = lanes 16-31
  const int l16    = lane & 15;
  const int tileN0 = blockIdx.x * (16 * NT);
  const int tileM  = (blockIdx.y * 8 + wave) * 16;

  const __bf16* aptr  = X + (size_t)(tileM + l16) * K;
  const __bf16* wbase = W + (size_t)(tileN0 + l16) * K;
  const int koff0 = half ? 8 : 0;
  const int koff1 = half ? 24 : 16;

  v8f acc[NT];
  #pragma unroll
  for (int t = 0; t < NT; ++t) {
    const float bv = bias[tileN0 + 16 * t + l16];
    #pragma unroll
    for (int i = 0; i < 8; ++i) acc[t][i] = bv;
  }

  for (int k = 0; k < K; k += 32) {
    const v8bf alo = *reinterpret_cast<const v8bf*>(aptr + k + koff0);
    const v8bf ahi = *reinterpret_cast<const v8bf*>(aptr + k + koff1);
    v16bf a;
    #pragma unroll
    for (int i = 0; i < 8; ++i) { a[i] = alo[i]; a[i + 8] = ahi[i]; }

    #pragma unroll
    for (int t = 0; t < NT; ++t) {
      const __bf16* wp = wbase + (size_t)(16 * t) * K;   // constant stride -> ioffset
      const v8bf blo = *reinterpret_cast<const v8bf*>(wp + k + koff0);
      const v8bf bhi = *reinterpret_cast<const v8bf*>(wp + k + koff1);
      v16bf bm;
      #pragma unroll
      for (int i = 0; i < 8; ++i) { bm[i] = blo[i]; bm[i + 8] = bhi[i]; }
      acc[t] = __builtin_amdgcn_wmma_f32_16x16x32_bf16(
                   /*neg_a=*/false, a, /*neg_b=*/false, bm,
                   /*c_mod=*/(short)0, acc[t], /*reuse_a=*/false, /*reuse_b=*/false);
    }
  }

  #pragma unroll
  for (int t = 0; t < NT; ++t) {
    const int col = tileN0 + 16 * t + l16;
    #pragma unroll
    for (int r = 0; r < 8; ++r) {
      Y[(size_t)(tileM + r + half * 8) * Cout + col] = acc[t][r];
    }
  }
}

// ---------------------------------------------------------------------------
// Per-channel sum / sum-of-squares over Y[M, C].  Thread pinned to a column,
// register accumulation over a 256-row chunk, then one f32 atomic per column.
// stats[0:C] = sum, stats[C:2C] = sumsq.
// ---------------------------------------------------------------------------
__global__ __launch_bounds__(256)
void fp_bn_stats(const float* __restrict__ Y, float* __restrict__ stats,
                 int M, int C, int rowsPerBlock) {
  const int r0 = blockIdx.x * rowsPerBlock;
  for (int c = threadIdx.x; c < C; c += 256) {
    float s = 0.0f, s2 = 0.0f;
    for (int r = r0; r < r0 + rowsPerBlock; ++r) {
      const float v = Y[(size_t)r * C + c];
      s += v;
      s2 += v * v;
    }
    atomicAdd(&stats[c], s);
    atomicAdd(&stats[C + c], s2);
  }
}

// ---------------------------------------------------------------------------
// BatchNorm (batch stats) + ReLU.  Out is bf16 (feeds next GEMM) or f32.
// ---------------------------------------------------------------------------
template <typename OutT>
__global__ __launch_bounds__(256)
void fp_bn_relu(const float* __restrict__ Y, const float* __restrict__ stats,
                const float* __restrict__ g, const float* __restrict__ beta,
                OutT* __restrict__ out, int M, int C) {
  const size_t i = (size_t)blockIdx.x * 256 + threadIdx.x;
  if (i >= (size_t)M * C) return;
  const int c = (int)(i % C);
  const float invM = 1.0f / (float)M;
  const float mu   = stats[c] * invM;
  const float var  = stats[C + c] * invM - mu * mu;
  const float rstd = rsqrtf(var + BN_EPS);
  float v = (Y[i] - mu) * rstd * g[c] + beta[c];
  v = fmaxf(v, 0.0f);
  out[i] = (OutT)v;
}

// ---------------------------------------------------------------------------
// Launch: interp/concat -> GEMM1 -> stats -> BN/ReLU -> GEMM2 -> stats ->
//         BN/ReLU -> out.  X2 aliases X1, Y2 aliases Y1 (dead by then).
// ---------------------------------------------------------------------------
extern "C" void kernel_launch(void* const* d_in, const int* in_sizes, int n_in,
                              void* d_out, int out_size, void* d_ws, size_t ws_size,
                              hipStream_t stream) {
  const float* xyz1    = (const float*)d_in[0];
  const float* xyz2    = (const float*)d_in[1];
  const float* points1 = (const float*)d_in[2];
  const float* points2 = (const float*)d_in[3];
  const float* w1      = (const float*)d_in[4];
  const float* b1      = (const float*)d_in[5];
  const float* g1      = (const float*)d_in[6];
  const float* beta1   = (const float*)d_in[7];
  const float* w2      = (const float*)d_in[8];
  const float* b2      = (const float*)d_in[9];
  const float* g2      = (const float*)d_in[10];
  const float* beta2   = (const float*)d_in[11];
  float* out = (float*)d_out;

  // Workspace layout (256B-aligned offsets)
  char* ws = (char*)d_ws;
  const size_t offX     = 0;                                   // bf16, 65536*384  (X2 aliases)
  const size_t offY     = offX + (size_t)PM * PK1 * 2;         // f32,  65536*256  (Y2 aliases)
  const size_t offW1    = offY + (size_t)PM * PO1 * 4;         // bf16, 256*384
  const size_t offW2    = offW1 + (size_t)PO1 * PK1 * 2;       // bf16, 128*256
  const size_t offStats = offW2 + (size_t)PO2 * PO1 * 2;       // f32,  2*256

  __bf16* X1    = (__bf16*)(ws + offX);
  __bf16* X2    = (__bf16*)(ws + offX);          // alias, written after X1 is dead
  float*  Y1    = (float*)(ws + offY);
  float*  Y2    = (float*)(ws + offY);           // alias, written after Y1 is dead
  __bf16* W1b   = (__bf16*)(ws + offW1);
  __bf16* W2b   = (__bf16*)(ws + offW2);
  float*  stats = (float*)(ws + offStats);

  // 0) weights -> bf16
  {
    const int n1 = PO1 * PK1, n2 = PO2 * PO1;
    fp_f32_to_bf16<<<(n1 + 255) / 256, 256, 0, stream>>>(w1, W1b, n1);
    fp_f32_to_bf16<<<(n2 + 255) / 256, 256, 0, stream>>>(w2, W2b, n2);
  }

  // 1) 3-NN interpolation + concat -> X1 [M, 384] bf16
  fp_interp_concat<<<dim3(PN / 256, PB), 256, 0, stream>>>(
      xyz1, xyz2, points1, points2, X1);

  // 2) GEMM1: Y1 = X1 * W1^T + b1   (WMMA bf16, f32 accumulate; 8 N-tiles/wave)
  fp_zero_f32<<<(2 * PO1 + 255) / 256, 256, 0, stream>>>(stats, 2 * PO1);
  fp_gemm_bf16_wmma<8, PK1><<<dim3(PO1 / 128, PM / 128), 256, 0, stream>>>(
      X1, W1b, b1, Y1, PM, PO1);

  // 3) BN stats + BN/ReLU -> X2 [M, 256] bf16
  fp_bn_stats<<<PM / 256, 256, 0, stream>>>(Y1, stats, PM, PO1, 256);
  {
    const size_t n = (size_t)PM * PO1;
    fp_bn_relu<__bf16><<<(unsigned)((n + 255) / 256), 256, 0, stream>>>(
        Y1, stats, g1, beta1, X2, PM, PO1);
  }

  // 4) GEMM2: Y2 = X2 * W2^T + b2
  fp_zero_f32<<<(2 * PO2 + 255) / 256, 256, 0, stream>>>(stats, 2 * PO2);
  fp_gemm_bf16_wmma<8, PO1><<<dim3(PO2 / 128, PM / 128), 256, 0, stream>>>(
      X2, W2b, b2, Y2, PM, PO2);

  // 5) BN stats + BN/ReLU -> out [M, 128] f32
  fp_bn_stats<<<PM / 256, 256, 0, stream>>>(Y2, stats, PM, PO2, 256);
  {
    const size_t n = (size_t)PM * PO2;
    fp_bn_relu<float><<<(unsigned)((n + 255) / 256), 256, 0, stream>>>(
        Y2, stats, g2, beta2, out, PM, PO2);
  }
}
